// MultiQueryAttention_10187662426351
// MI455X (gfx1250) — matrix operand order repaired and checked by reading, not compiled, not run on hardware
//
#include <hip/hip_runtime.h>
#include <hip/hip_bf16.h>
#include <math.h>
#include <cstdint>

typedef __attribute__((ext_vector_type(16))) __bf16 v16bf;
typedef __attribute__((ext_vector_type(8)))  __bf16 v8bf;
typedef __attribute__((ext_vector_type(4)))  __bf16 v4bf;
typedef __attribute__((ext_vector_type(8)))  float  v8f;
typedef __attribute__((ext_vector_type(4)))  uint32_t u32x4;
typedef __attribute__((ext_vector_type(8)))  uint32_t u32x8;

constexpr int BB = 2, SS = 2048, DD = 2048, HH = 16, HDD = 128;
constexpr int BSROWS = BB * SS;                    // 4096
constexpr float ATT_SCALE = 0.0883883476483184f;   // 1/sqrt(128)

static __device__ inline v8f vzero8() {
  v8f z;
#pragma unroll
  for (int i = 0; i < 8; ++i) z[i] = 0.0f;
  return z;
}

static __device__ inline v8f wmma_bf16(v16bf a, v16bf b, v8f c) {
  return __builtin_amdgcn_wmma_f32_16x16x32_bf16(
      false, a, false, b, (short)0, c, false, false);
}

// A 16x32 bf16 fragment from row-major bf16 (K contiguous).
static __device__ inline v16bf load_A_bf(const __bf16* rowp, int half) {
  v8bf lo = *(const v8bf*)(rowp + half * 16);
  v8bf hi = *(const v8bf*)(rowp + half * 16 + 8);
  return __builtin_shufflevector(lo, hi, 0, 1, 2, 3, 4, 5, 6, 7,
                                 8, 9, 10, 11, 12, 13, 14, 15);
}

// B 32x16 bf16 fragment, K contiguous per n=lane&15 row (32B load).
static __device__ inline v16bf load_B_bf(const __bf16* p) {
  return *(const v16bf*)p;
}

static __device__ inline void async_load_b128(uint32_t lds_off, const void* gp) {
  // gfx1250 async copy global -> LDS (ASYNCcnt)
  asm volatile("global_load_async_to_lds_b128 %0, %1, off"
               :: "v"(lds_off), "v"(gp) : "memory");
}

// gfx1250 Tensor Data Mover: 2D tile load via D# groups 0/1 in SGPRs
// (VADDR2/VADDR3 = NULL -> 2-group form, tensors up to 2D). TENSORcnt.
static __device__ inline void tdm_load_2d(u32x4 g0, u32x8 g1) {
  asm volatile("tensor_load_to_lds %0, %1" :: "s"(g0), "s"(g1) : "memory");
}

// ---- f32 -> bf16 elementwise convert (x4 per thread) ----------------------
__global__ void __launch_bounds__(256)
cvt_f32_bf16(const float4* __restrict__ s, v4bf* __restrict__ d, int n4) {
  int i = blockIdx.x * 256 + threadIdx.x;
  if (i < n4) {
    float4 f = s[i];
    v4bf o;
    o[0] = (__bf16)f.x; o[1] = (__bf16)f.y;
    o[2] = (__bf16)f.z; o[3] = (__bf16)f.w;
    d[i] = o;
  }
}

// ---- f32 [R][C] -> bf16 [C][R] tiled transpose+convert ---------------------
__global__ void __launch_bounds__(256)
transpose_cvt(const float* __restrict__ src, __bf16* __restrict__ dst,
              int R, int C) {
  __shared__ float tile[32][33];
  const int tx = threadIdx.x, ty = threadIdx.y;       // 32 x 8
  const int c0 = blockIdx.x * 32, r0 = blockIdx.y * 32;
#pragma unroll
  for (int i = 0; i < 32; i += 8)
    tile[ty + i][tx] = src[(size_t)(r0 + ty + i) * C + c0 + tx];
  __syncthreads();
#pragma unroll
  for (int i = 0; i < 32; i += 8)
    dst[(size_t)(c0 + ty + i) * R + r0 + tx] = (__bf16)tile[tx][ty + i];
}

// ---- GEMM + bias: C[M,N] = A[M,K](bf16) @ WT[N,K](bf16)^T + bias -----------
// 4 waves/block, 32x64 tile per wave (2x4 WMMA tiles): 8 wmma per 6 fragment
// loads per K-chunk, all operands as b128 vector loads.
template <bool BF16OUT>
__global__ void __launch_bounds__(128)
gemm_bias(const __bf16* __restrict__ A, const __bf16* __restrict__ WT,
          const float* __restrict__ bias, void* __restrict__ Cout,
          int M, int N, int K) {
  (void)M;
  const int lane = threadIdx.x & 31, wid = threadIdx.x >> 5;
  const int half = lane >> 4, l15 = lane & 15;
  const int gw = blockIdx.x * 4 + wid;
  const int ntn = N >> 6;
  const int tm = gw / ntn, tn = gw % ntn;
  const int rows0 = tm * 32, cols0 = tn * 64;

  v8f acc0[4], acc1[4];
#pragma unroll
  for (int j = 0; j < 4; ++j) { acc0[j] = vzero8(); acc1[j] = vzero8(); }

  const __bf16* a0p = A + (size_t)(rows0 + l15) * K;
  const __bf16* a1p = a0p + (size_t)16 * K;
  const __bf16* bp0 = WT + (size_t)(cols0 + l15) * K;

  for (int k0 = 0; k0 < K; k0 += 32) {
    if (k0 + 32 < K) {
      __builtin_prefetch(a0p + k0 + 32, 0, 0);
      __builtin_prefetch(bp0 + k0 + 32, 0, 0);
    }
    v16bf a0 = load_A_bf(a0p + k0, half);
    v16bf a1 = load_A_bf(a1p + k0, half);
#pragma unroll
    for (int j = 0; j < 4; ++j) {
      v16bf bf = load_B_bf(bp0 + (size_t)(j * 16) * K + k0 + half * 16);
      acc0[j] = wmma_bf16(a0, bf, acc0[j]);
      acc1[j] = wmma_bf16(a1, bf, acc1[j]);
    }
  }

  const size_t rbase = (size_t)(rows0 + half * 8) * N;
#pragma unroll
  for (int j = 0; j < 4; ++j) {
    const int col = cols0 + j * 16 + l15;
    const float bvv = bias[col];
    if constexpr (BF16OUT) {
      __bf16* C = (__bf16*)Cout;
#pragma unroll
      for (int r = 0; r < 8; ++r) {
        C[rbase + (size_t)r * N + col]        = (__bf16)(acc0[j][r] + bvv);
        C[rbase + (size_t)(r + 16) * N + col] = (__bf16)(acc1[j][r] + bvv);
      }
    } else {
      float* C = (float*)Cout;
#pragma unroll
      for (int r = 0; r < 8; ++r) {
        C[rbase + (size_t)r * N + col]        = acc0[j][r] + bvv;
        C[rbase + (size_t)(r + 16) * N + col] = acc1[j][r] + bvv;
      }
    }
  }
}

// ---- Fused MQA attention ---------------------------------------------------
// Block = 4 waves = 4 heads of one (b, s-tile): identical causal trip count,
// identical K/V stream. K chunk staged by the Tensor Data Mover (one
// tensor_load_to_lds per chunk, TENSORcnt), V^T chunk by per-lane async
// copies (ASYNCcnt); both double buffered in LDS, shared by all heads.
__global__ void __launch_bounds__(128)
mqa_attention(const __bf16* __restrict__ Q, const __bf16* __restrict__ Kbf,
              const __bf16* __restrict__ VT, __bf16* __restrict__ X) {
  const int tid = threadIdx.x;
  const int lane = tid & 31, wid = tid >> 5;
  const int half = lane >> 4, l15 = lane & 15;

  int idx = blockIdx.x;
  const int st = idx & 127; idx >>= 7;     // s-tile
  const int hg = idx & 3;                  // head group
  const int b  = idx >> 2;
  const int h = hg * 4 + wid;
  const int s_base = st * 16;

  __shared__ alignas(32) __bf16 kbuf[2][32 * 128];  // [t_local][d]
  __shared__ alignas(32) __bf16 vbuf[2][128 * 32];  // [d][t_local]
  __shared__ alignas(32) __bf16 pls[4][16 * 32];    // per-wave P staging

  const uint32_t kbase = (uint32_t)(uintptr_t)&kbuf[0][0];
  const uint32_t vbase = (uint32_t)(uintptr_t)&vbuf[0][0];

  const __bf16* Kb  = Kbf + (size_t)b * SS * HDD;   // [S][HD]
  const __bf16* VTb = VT + (size_t)b * HDD * SS;    // [HD][S]
  __bf16* pbuf = &pls[wid][0];

  // Q fragments: 16 rows x 128, four K=32 chunks (loaded once).
  const __bf16* qrow = Q + ((size_t)b * SS + s_base + l15) * DD + (size_t)h * HDD;
  v16bf qf[4];
#pragma unroll
  for (int c = 0; c < 4; ++c) qf[c] = load_A_bf(qrow + c * 32, half);

  v8f acc[8];
#pragma unroll
  for (int j = 0; j < 8; ++j) acc[j] = vzero8();
  float mrow[8], lrow[8];
#pragma unroll
  for (int r = 0; r < 8; ++r) { mrow[r] = -__builtin_inff(); lrow[r] = 0.0f; }

  const int nchunk = (s_base + 47) >> 5;   // causal 32-key chunks

  // K chunk: 2D TDM tile  (tensor [S][HD] bf16, tile 32x128 at row t0).
  auto issueK = [&](int ch, int bufi) {
    const uint64_t ga = (uint64_t)(uintptr_t)(Kb + (size_t)ch * 32 * HDD);
    u32x4 g0;
    g0[0] = 1u;                                           // count=1 (valid D#)
    g0[1] = kbase + (uint32_t)bufi * 8192u;               // lds_addr
    g0[2] = (uint32_t)ga;                                 // global_addr[31:0]
    g0[3] = (uint32_t)((ga >> 32) & 0x01FFFFFFu) | (2u << 30);  // addr hi, type=2
    u32x8 g1;
    g1[0] = 1u << 16;                                     // data_size=2B
    g1[1] = (uint32_t)HDD << 16;                          // tensor_dim0 lo
    g1[2] = (uint32_t)SS << 16;                           // dim0 hi | tensor_dim1 lo
    g1[3] = (uint32_t)HDD << 16;                          // dim1 hi | tile_dim0=128
    g1[4] = 32u;                                          // tile_dim1=32
    g1[5] = (uint32_t)HDD;                                // tensor_dim0_stride
    g1[6] = 0u;
    g1[7] = 0u;
    tdm_load_2d(g0, g1);
  };

  // V^T chunk: 128 rows x 32 cols, per-lane 16B async copies (4 per wave).
  auto issueV = [&](int ch, int bufi) {
    const int t0 = ch * 32;
#pragma unroll
    for (int i = 0; i < 4; ++i) {
      const int seg = tid + i * 128;               // 16B segment id (0..511)
      const int dr = seg >> 2, part = seg & 3;     // V^T row d, 8-elem part
      const char* vc = (const char*)(VTb + (size_t)dr * SS + t0 + part * 8);
      async_load_b128(vbase + (uint32_t)bufi * 8192 + (uint32_t)seg * 16, vc);
    }
  };

  if (wid == 0) issueK(0, 0);
  issueV(0, 0);

  for (int ch = 0; ch < nchunk; ++ch) {
    const int bufi = ch & 1;
    if (ch + 1 < nchunk) {
      if (wid == 0) issueK(ch + 1, bufi ^ 1);
      issueV(ch + 1, bufi ^ 1);
      __builtin_amdgcn_s_wait_tensorcnt(1);              // current K tile done
      asm volatile("s_wait_asynccnt 0x4" ::: "memory");  // current V tile done
    } else {
      __builtin_amdgcn_s_wait_tensorcnt(0);
      asm volatile("s_wait_asynccnt 0x0" ::: "memory");
    }
    __syncthreads();

    const int t0 = ch * 32;
    const __bf16* kl = &kbuf[bufi][0];
    const __bf16* vl = &vbuf[bufi][0];

    // ---- scores: two 16x16 tiles over d = 0..127 ----
    v8f s0 = vzero8(), s1 = vzero8();
#pragma unroll
    for (int c = 0; c < 4; ++c) {
      v16bf kb0 = load_B_bf(kl + l15 * 128 + c * 32 + half * 16);
      v16bf kb1 = load_B_bf(kl + (l15 + 16) * 128 + c * 32 + half * 16);
      s0 = wmma_bf16(qf[c], kb0, s0);
      s1 = wmma_bf16(qf[c], kb1, s1);
    }

    // ---- scale, causal mask, online softmax ----
#pragma unroll
    for (int r = 0; r < 8; ++r) {
      const int srow = s_base + r + half * 8;
      const int tc0 = t0 + l15, tc1 = tc0 + 16;
      float v0 = (tc0 > srow) ? -__builtin_inff() : s0[r] * ATT_SCALE;
      float v1 = (tc1 > srow) ? -__builtin_inff() : s1[r] * ATT_SCALE;
      float cur = fmaxf(v0, v1);
#pragma unroll
      for (int msk = 1; msk < 16; msk <<= 1)
        cur = fmaxf(cur, __shfl_xor(cur, msk, 32));
      const float mnew  = fmaxf(mrow[r], cur);
      const float alpha = __expf(mrow[r] - mnew);
      mrow[r] = mnew;
      const float e0 = __expf(v0 - mnew);
      const float e1 = __expf(v1 - mnew);
      float rs = e0 + e1;
#pragma unroll
      for (int msk = 1; msk < 16; msk <<= 1)
        rs += __shfl_xor(rs, msk, 32);
      lrow[r] = lrow[r] * alpha + rs;
#pragma unroll
      for (int j = 0; j < 8; ++j) acc[j][r] *= alpha;
      const int prow = r + half * 8;
      pbuf[prow * 32 + l15]      = (__bf16)e0;
      pbuf[prow * 32 + 16 + l15] = (__bf16)e1;
    }
    __builtin_amdgcn_wave_barrier();
    asm volatile("s_wait_dscnt 0x0" ::: "memory");  // P store -> cross-lane load

    v16bf pf = load_A_bf(pbuf + l15 * 32, half);    // P as 16x32 A-fragment
    __builtin_amdgcn_wave_barrier();

    // ---- O += P(16x32) @ V(32x128), V^T fragments contiguous in LDS ----
#pragma unroll
    for (int j = 0; j < 8; ++j) {
      v16bf vf = load_B_bf(vl + (j * 16 + l15) * 32 + half * 16);
      acc[j] = wmma_bf16(pf, vf, acc[j]);
    }
    __syncthreads();   // everyone done with buf before it is refilled
  }

  // ---- normalize, store in straight-reshape layout (one contiguous row) ----
  __bf16* xr = X + ((size_t)b * SS + (size_t)h * (SS / 16) + st) * DD;
#pragma unroll
  for (int r = 0; r < 8; ++r) {
    const float inv = 1.0f / lrow[r];
    const int colbase = (r + half * 8) * HDD;
#pragma unroll
    for (int j = 0; j < 8; ++j)
      xr[colbase + j * 16 + l15] = (__bf16)(acc[j][r] * inv);
  }
}

// ---- Host-side orchestration ----------------------------------------------
extern "C" void kernel_launch(void* const* d_in, const int* in_sizes, int n_in,
                              void* d_out, int out_size, void* d_ws, size_t ws_size,
                              hipStream_t stream) {
  (void)in_sizes; (void)n_in; (void)out_size; (void)ws_size;
  const float* input = (const float*)d_in[0];
  const float* kv    = (const float*)d_in[1];
  const float* Wq    = (const float*)d_in[2];
  const float* bq    = (const float*)d_in[3];
  const float* Wk    = (const float*)d_in[4];
  const float* bk    = (const float*)d_in[5];
  const float* Wv    = (const float*)d_in[6];
  const float* bv    = (const float*)d_in[7];
  const float* Wo    = (const float*)d_in[8];
  const float* bo    = (const float*)d_in[9];
  float* out = (float*)d_out;

  char* w = (char*)d_ws;
  __bf16* in_bf = (__bf16*)w; w += (size_t)BSROWS * DD * 2;
  __bf16* kv_bf = (__bf16*)w; w += (size_t)BSROWS * DD * 2;
  __bf16* WqT   = (__bf16*)w; w += (size_t)DD * DD * 2;
  __bf16* WkT   = (__bf16*)w; w += (size_t)HDD * DD * 2;
  __bf16* WvT   = (__bf16*)w; w += (size_t)HDD * DD * 2;
  __bf16* WoT   = (__bf16*)w; w += (size_t)DD * DD * 2;
  __bf16* Qbf   = (__bf16*)w; w += (size_t)BSROWS * DD * 2;
  __bf16* Kbf   = (__bf16*)w; w += (size_t)BSROWS * HDD * 2;
  float*  Vtmp  = (float*)w;  w += (size_t)BSROWS * HDD * 4;
  __bf16* VTbf  = (__bf16*)w; w += (size_t)BB * HDD * SS * 2;
  __bf16* Xbf   = (__bf16*)w;

  const int nAct4 = (BSROWS * DD) / 4;
  cvt_f32_bf16<<<nAct4 / 256, 256, 0, stream>>>((const float4*)input, (v4bf*)in_bf, nAct4);
  cvt_f32_bf16<<<nAct4 / 256, 256, 0, stream>>>((const float4*)kv, (v4bf*)kv_bf, nAct4);

  const dim3 tb(32, 8);
  transpose_cvt<<<dim3(DD / 32, DD / 32), tb, 0, stream>>>(Wq, WqT, DD, DD);
  transpose_cvt<<<dim3(HDD / 32, DD / 32), tb, 0, stream>>>(Wk, WkT, DD, HDD);
  transpose_cvt<<<dim3(HDD / 32, DD / 32), tb, 0, stream>>>(Wv, WvT, DD, HDD);
  transpose_cvt<<<dim3(DD / 32, DD / 32), tb, 0, stream>>>(Wo, WoT, DD, DD);

  const dim3 blk(128);
  const int blocksBig = ((BSROWS / 32) * (DD / 64)) / 4;    // 1024
  const int blocksKV  = ((BSROWS / 32) * (HDD / 64)) / 4;   // 64
  gemm_bias<true><<<blocksBig, blk, 0, stream>>>(in_bf, WqT, bq, Qbf, BSROWS, DD, DD);
  gemm_bias<true><<<blocksKV, blk, 0, stream>>>(kv_bf, WkT, bk, Kbf, BSROWS, HDD, DD);
  gemm_bias<false><<<blocksKV, blk, 0, stream>>>(kv_bf, WvT, bv, Vtmp, BSROWS, HDD, DD);

  // V [S][HD] f32 -> V^T [HD][S] bf16, per batch
  for (int b = 0; b < BB; ++b)
    transpose_cvt<<<dim3(HDD / 32, SS / 32), tb, 0, stream>>>(
        Vtmp + (size_t)b * SS * HDD, VTbf + (size_t)b * HDD * SS, SS, HDD);

  const int blocksA = BB * 4 * (SS / 16);                   // 1024
  mqa_attention<<<blocksA, blk, 0, stream>>>(Qbf, Kbf, VTbf, Xbf);

  gemm_bias<false><<<blocksBig, blk, 0, stream>>>(Xbf, WoT, bo, out, BSROWS, DD, DD);
}
